// GraphAttentionLayer_60730837565928
// MI455X (gfx1250) — matrix operand order repaired and checked by reading, not compiled, not run on hardware
//
#include <hip/hip_runtime.h>

typedef __attribute__((ext_vector_type(16))) _Float16 v16h;
typedef __attribute__((ext_vector_type(8)))  float    v8f;

#define GAT_B    8
#define GAT_N    4096
#define GAT_FIN  128
#define GAT_FOUT 64
#define ADJ_W    (GAT_N / 32)      /* 128 u32 words per adjacency row */
#define NEG_BIG  (-1e9f)
#define INV_SQRT_D 0.125f          /* 1/sqrt(64) */

// ---------------------------------------------------------------------------
// 16-lane-row all-reduce max. DPP16 row rotates (row = 16 lanes) stay inside
// each half-wave group, which matches the WMMA C/D row grouping exactly.
// ---------------------------------------------------------------------------
#if __has_builtin(__builtin_amdgcn_update_dpp)
#define ROW_ROR(x, N)                                                         \
    __builtin_bit_cast(float, __builtin_amdgcn_update_dpp(                    \
        0, __builtin_bit_cast(int, (x)), 0x120 | (N), 0xF, 0xF, true))
__device__ __forceinline__ float rowmax16(float x) {
    x = fmaxf(x, ROW_ROR(x, 1));
    x = fmaxf(x, ROW_ROR(x, 2));
    x = fmaxf(x, ROW_ROR(x, 4));
    x = fmaxf(x, ROW_ROR(x, 8));
    return x;
}
#else
__device__ __forceinline__ float rowmax16(float x) {
    x = fmaxf(x, __shfl_xor(x, 1));
    x = fmaxf(x, __shfl_xor(x, 2));
    x = fmaxf(x, __shfl_xor(x, 4));
    x = fmaxf(x, __shfl_xor(x, 8));
    return x;
}
#endif

__device__ __forceinline__ float rowsum16(float x) {
    x += __shfl_xor(x, 1);
    x += __shfl_xor(x, 2);
    x += __shfl_xor(x, 4);
    x += __shfl_xor(x, 8);
    return x;
}

// ---------------------------------------------------------------------------
// Kernel 0: pack adj[N][N] (int32 0/1) into bitmask words adj_bits[N][N/32].
// 67 MB -> 2 MB; read once, L2-resident afterwards.
// ---------------------------------------------------------------------------
__global__ void gat_pack_adj(const int* __restrict__ adj,
                             unsigned* __restrict__ bits) {
    const int row = blockIdx.x;        // 0..N-1
    const int w   = threadIdx.x;       // 0..127
    const int* p  = adj + (size_t)row * GAT_N + w * 32;
    unsigned word = 0u;
#pragma unroll
    for (int i = 0; i < 32; ++i)
        word |= (p[i] > 0 ? 1u : 0u) << i;
    bits[(size_t)row * ADJ_W + w] = word;
}

// ---------------------------------------------------------------------------
// Kernel 1: Wh = h @ W, emitted as f16 twice:
//   wh  [B][N][64]  row-major   (Q rows / K columns for the score WMMAs)
//   wht [B][64][N]  transposed  (V B-operand for the PV WMMAs: contiguous K)
// Tiled: one 256-thread block per 64 rows. W (32KB) + h-block (32KB, bank-
// padded) staged in LDS; transpose goes through a padded LDS tile so the wht
// writes are contiguous 32B runs per thread (no scattered 2B stores).
// ---------------------------------------------------------------------------
__global__ __launch_bounds__(256) void gat_compute_wh(
        const float* __restrict__ h, const float* __restrict__ W,
        _Float16* __restrict__ wh, _Float16* __restrict__ wht) {
    const int blk = blockIdx.x;                  // 0 .. B*N/64 - 1
    const int b   = blk / (GAT_N / 64);
    const int n0  = (blk % (GAT_N / 64)) * 64;
    const int tid = threadIdx.x;

    __shared__ float     Ws[GAT_FIN][GAT_FOUT];  // 32 KB
    __shared__ float     Hs[64][GAT_FIN + 2];    // ~33 KB, pad kills bank conflicts
    __shared__ _Float16  Ts[GAT_FOUT][64 + 8];   // ~9 KB, padded transpose tile

    for (int i = tid; i < GAT_FIN * GAT_FOUT; i += 256)
        (&Ws[0][0])[i] = W[i];
    const float* hblk = h + ((size_t)b * GAT_N + n0) * GAT_FIN;
    for (int i = tid; i < 64 * GAT_FIN; i += 256)
        Hs[i >> 7][i & 127] = hblk[i];
    __syncthreads();

    const int r  = tid >> 2;               // row within tile 0..63
    const int o0 = (tid & 3) * 16;         // feature sub-block 0/16/32/48
    float acc[16];
#pragma unroll
    for (int j = 0; j < 16; ++j) acc[j] = 0.0f;
    for (int f = 0; f < GAT_FIN; ++f) {
        const float hv = Hs[r][f];
#pragma unroll
        for (int j = 0; j < 16; ++j)
            acc[j] = fmaf(hv, Ws[f][o0 + j], acc[j]);
    }

    _Float16* whrow = wh + ((size_t)b * GAT_N + n0 + r) * GAT_FOUT + o0;
#pragma unroll
    for (int j = 0; j < 16; ++j) {
        const _Float16 v = (_Float16)acc[j];
        whrow[j]       = v;       // coalesced 32B run per thread
        Ts[o0 + j][r]  = v;       // LDS transpose
    }
    __syncthreads();

    // contiguous transposed write: thread -> 16 halves (32B) of one feature row
    const int of = tid >> 2;
    const int c0 = (tid & 3) * 16;
    _Float16*       dst = wht + ((size_t)b * GAT_FOUT + of) * GAT_N + n0 + c0;
    const _Float16* src = &Ts[of][c0];
#pragma unroll
    for (int j = 0; j < 16; ++j) dst[j] = src[j];
}

// ---------------------------------------------------------------------------
// Kernel 2: fused flash-attention GAT body + ELU.
// Block = 128 threads = 4 independent waves; each wave owns a 16-row tile of
// one batch and streams keys in chunks of 32 with online softmax.
// Per chunk: 4x wmma (scores, K=64 feats) + mask + DPP max-reduce + LDS P
// transpose + 4x wmma (P.V). Row-sum denominator is kept per-lane partial and
// reduced once at the end.
// ---------------------------------------------------------------------------
__global__ __launch_bounds__(128) void gat_flash(
        const _Float16* __restrict__ wh,
        const _Float16* __restrict__ wht,
        const unsigned* __restrict__ bits,
        float* __restrict__ out) {

    const int lane = threadIdx.x & 31;
    const int wave = threadIdx.x >> 5;
    const int b    = blockIdx.x >> 6;                       // grid = B * 64
    const int tile = ((blockIdx.x & 63) << 2) + wave;       // 0..255
    const int rowbase = tile << 4;
    const int lo = lane & 15;          // column / row-within-tile selector
    const int hi = lane >> 4;          // half-wave selector

    __shared__ _Float16 pstage[4][16 * 32];
    _Float16* ps = pstage[wave];

    const _Float16* whB  = wh  + (size_t)b * GAT_N * GAT_FOUT;
    const _Float16* whtB = wht + (size_t)b * GAT_FOUT * GAT_N;

    // ---- Q A-operands (fixed for the whole loop), contiguous k = 16*hi + s
    const _Float16* qrow = whB + (size_t)(rowbase + lo) * GAT_FOUT;
    v16h a0, a1;
    {
        const int f0 = hi * 16;
#pragma unroll
        for (int i = 0; i < 8; ++i) {
            a0[i]     = qrow[f0 + i];
            a0[8 + i] = qrow[8 + f0 + i];
            a1[i]     = qrow[32 + f0 + i];
            a1[8 + i] = qrow[40 + f0 + i];
        }
    }

    // Online-softmax state. Element v of lane L is row (v + hi*8); lrow holds
    // this lane's partial denominator (reduced across the 16-lane row at end).
    float mrow[8], lrow[8];
    v8f acc0 = {}, acc1 = {}, acc2 = {}, acc3 = {};
#pragma unroll
    for (int v = 0; v < 8; ++v) { mrow[v] = -1e30f; lrow[v] = 0.0f; }

    const unsigned* bitbase = bits + (size_t)rowbase * ADJ_W;

    for (int kb = 0; kb < GAT_N; kb += 32) {
        // ---- score B-operands: lane = key column, 16 contiguous feature halves
        const _Float16* krow0 = whB + (size_t)(kb +      lo) * GAT_FOUT + hi * 16;
        const _Float16* krow1 = whB + (size_t)(kb + 16 + lo) * GAT_FOUT + hi * 16;
        if (kb + 32 < GAT_N) {                     // cover L2->L0 latency
            __builtin_prefetch(krow0 + 32 * GAT_FOUT, 0, 3);
            __builtin_prefetch(krow1 + 32 * GAT_FOUT, 0, 3);
        }
        v16h b00, b01, b10, b11;
#pragma unroll
        for (int i = 0; i < 16; ++i) {
            b00[i] = krow0[i];      b01[i] = krow0[32 + i];
            b10[i] = krow1[i];      b11[i] = krow1[32 + i];
        }
        v8f s0 = {}, s1 = {};
        s0 = __builtin_amdgcn_wmma_f32_16x16x32_f16(false, a0, false, b00, (short)0, s0, false, false);
        s1 = __builtin_amdgcn_wmma_f32_16x16x32_f16(false, a0, false, b10, (short)0, s1, false, false);
        s0 = __builtin_amdgcn_wmma_f32_16x16x32_f16(false, a1, false, b01, (short)0, s0, false, false);
        s1 = __builtin_amdgcn_wmma_f32_16x16x32_f16(false, a1, false, b11, (short)0, s1, false, false);

        // ---- mask via adjacency bits + online softmax (DPP row all-reduce)
        float p0[8], p1[8], fac[8];
#pragma unroll
        for (int v = 0; v < 8; ++v) {
            const int m = v + hi * 8;
            const unsigned wb = bitbase[(size_t)m * ADJ_W + (kb >> 5)];
            const float sc0 = ((wb >> lo)        & 1u) ? s0[v] * INV_SQRT_D : NEG_BIG;
            const float sc1 = ((wb >> (16 + lo)) & 1u) ? s1[v] * INV_SQRT_D : NEG_BIG;
            const float mn  = fmaxf(mrow[v], rowmax16(fmaxf(sc0, sc1)));
            const float e0  = __expf(sc0 - mn);
            const float e1  = __expf(sc1 - mn);
            const float f   = __expf(mrow[v] - mn);
            p0[v] = e0; p1[v] = e1; fac[v] = f;
            lrow[v] = lrow[v] * f + (e0 + e1);   // per-lane partial denominator
            mrow[v] = mn;
        }
#pragma unroll
        for (int v = 0; v < 8; ++v) {
            acc0[v] *= fac[v]; acc1[v] *= fac[v];
            acc2[v] *= fac[v]; acc3[v] *= fac[v];
        }

        // ---- P: C/D layout -> A-operand layout via per-wave LDS round trip
#pragma unroll
        for (int v = 0; v < 8; ++v) {
            const int m = v + hi * 8;
            ps[m * 32 + lo]      = (_Float16)p0[v];
            ps[m * 32 + 16 + lo] = (_Float16)p1[v];
        }
        __builtin_amdgcn_wave_barrier();
        asm volatile("" ::: "memory");      // keep DS loads after DS stores
        v16h pa;
#pragma unroll
        for (int i = 0; i < 8; ++i) {
            pa[i]     = ps[lo * 32 + hi * 16 + i];
            pa[8 + i] = ps[lo * 32 + hi * 16 + 8 + i];
        }
        __builtin_amdgcn_wave_barrier();

        // ---- P (16x32) x V (32x64): V B-operand from transposed Wh, so each
        // lane reads 16 contiguous key halves of one feature column.
        {
            const _Float16* vr0 = whtB + (size_t)(lo)      * GAT_N + kb + hi * 16;
            const _Float16* vr1 = whtB + (size_t)(16 + lo) * GAT_N + kb + hi * 16;
            const _Float16* vr2 = whtB + (size_t)(32 + lo) * GAT_N + kb + hi * 16;
            const _Float16* vr3 = whtB + (size_t)(48 + lo) * GAT_N + kb + hi * 16;
            if (kb + 32 < GAT_N) {
                __builtin_prefetch(vr0 + 32, 0, 3);
                __builtin_prefetch(vr1 + 32, 0, 3);
                __builtin_prefetch(vr2 + 32, 0, 3);
                __builtin_prefetch(vr3 + 32, 0, 3);
            }
            v16h vb0, vb1, vb2, vb3;
#pragma unroll
            for (int i = 0; i < 16; ++i) {
                vb0[i] = vr0[i]; vb1[i] = vr1[i];
                vb2[i] = vr2[i]; vb3[i] = vr3[i];
            }
            acc0 = __builtin_amdgcn_wmma_f32_16x16x32_f16(false, pa, false, vb0, (short)0, acc0, false, false);
            acc1 = __builtin_amdgcn_wmma_f32_16x16x32_f16(false, pa, false, vb1, (short)0, acc1, false, false);
            acc2 = __builtin_amdgcn_wmma_f32_16x16x32_f16(false, pa, false, vb2, (short)0, acc2, false, false);
            acc3 = __builtin_amdgcn_wmma_f32_16x16x32_f16(false, pa, false, vb3, (short)0, acc3, false, false);
        }
    }

    // ---- finalize: reduce the deferred denominator, 1/l scaling + ELU
    float* orow = out + ((size_t)b * GAT_N + rowbase) * GAT_FOUT;
#pragma unroll
    for (int v = 0; v < 8; ++v) {
        const int m = v + hi * 8;
        const float inv = 1.0f / rowsum16(lrow[v]);
        float x0 = acc0[v] * inv, x1 = acc1[v] * inv;
        float x2 = acc2[v] * inv, x3 = acc3[v] * inv;
        x0 = (x0 > 0.0f) ? x0 : (__expf(x0) - 1.0f);
        x1 = (x1 > 0.0f) ? x1 : (__expf(x1) - 1.0f);
        x2 = (x2 > 0.0f) ? x2 : (__expf(x2) - 1.0f);
        x3 = (x3 > 0.0f) ? x3 : (__expf(x3) - 1.0f);
        float* o = orow + (size_t)m * GAT_FOUT + lo;
        o[0]  = x0;
        o[16] = x1;
        o[32] = x2;
        o[48] = x3;
    }
}

// ---------------------------------------------------------------------------
// Host-side launcher. Workspace layout (needs 10 MB):
//   [0, 4MB)   wh   f16  [B][N][64]
//   [4, 8MB)   wht  f16  [B][64][N]
//   [8,10MB)   bits u32  [N][N/32]
// ---------------------------------------------------------------------------
extern "C" void kernel_launch(void* const* d_in, const int* in_sizes, int n_in,
                              void* d_out, int out_size, void* d_ws, size_t ws_size,
                              hipStream_t stream) {
    (void)in_sizes; (void)n_in; (void)out_size; (void)ws_size;
    const float* h   = (const float*)d_in[0];
    const float* W   = (const float*)d_in[1];
    const int*   adj = (const int*)d_in[2];
    float*       out = (float*)d_out;

    char* ws = (char*)d_ws;
    _Float16* wh   = (_Float16*)(ws);
    _Float16* wht  = (_Float16*)(ws + (size_t)GAT_B * GAT_N * GAT_FOUT * sizeof(_Float16));
    unsigned* bits = (unsigned*)(ws + (size_t)GAT_B * GAT_N * GAT_FOUT * sizeof(_Float16) * 2);

    gat_pack_adj  <<<GAT_N,               128, 0, stream>>>(adj, bits);
    gat_compute_wh<<<GAT_B * (GAT_N / 64), 256, 0, stream>>>(h, W, wh, wht);
    gat_flash     <<<GAT_B * (GAT_N / 64), 128, 0, stream>>>(wh, wht, bits, out);
}
